// HNNLayer_53102975647843
// MI455X (gfx1250) — compile-verified, matrix operand order checked by reading, not run on hardware
//
#include <hip/hip_runtime.h>

typedef __attribute__((ext_vector_type(16))) _Float16 v16h;
typedef __attribute__((ext_vector_type(8)))  _Float16 v8h;
typedef __attribute__((ext_vector_type(8)))  float    v8f;

constexpr int DIM = 256;   // feature dim
constexpr int TM  = 64;    // rows per block

__device__ __forceinline__ float wred(float x) {
#pragma unroll
  for (int o = 16; o > 0; o >>= 1) x += __shfl_xor(x, o, 32);
  return x;
}

// ---------------------------------------------------------------------------
// Pre-shuffle W (f32, row-major DxD) into f16 WMMA B-fragments in d_ws.
// B fragment layout (16-bit, 32x16 K x N, wave32), mirroring the ISA A layout
// with lanes carrying N:  lane<16: i=0..7 -> K 0..7,  i=8..15 -> K 16..23
//                         lane>=16: i=0..7 -> K 8..15, i=8..15 -> K 24..31
// B[k][n] = W[n][k]  (since GEMM is x @ W^T).
// Fragment (jt,ks) is 32 lanes x 16 halves contiguous (1 KB); 16*8 fragments.
// ---------------------------------------------------------------------------
__global__ void prep_w_kernel(const float* __restrict__ W,
                              _Float16* __restrict__ wf) {
  int idx = blockIdx.x * 256 + threadIdx.x;    // 65536 elements
  int j = idx >> 8;                            // output col n (row of W)
  int k = idx & 255;                           // reduction index
  int jt = j >> 4, ks = k >> 5, ko = k & 31;
  int hi   = (ko >> 3) & 1;
  int lane = (j & 15) + (hi << 4);
  int i    = (ko & 7) + (((ko >> 4) & 1) << 3);
  wf[((jt * 8 + ks) * 32 + lane) * 16 + i] = (_Float16)W[idx];
}

// ---------------------------------------------------------------------------
// Fused HNN layer. 256 threads = 8 wave32s. 64 rows per block.
// LDS (64 KB) is a union:  phase A/B: A-fragments f16 (32 KB)
//                          phase C  : GEMM result G f32 (64 KB)
// ---------------------------------------------------------------------------
__launch_bounds__(256)
__global__ void hnn_fused_kernel(const float* __restrict__ H,
                                 const float* __restrict__ bias,
                                 const float* __restrict__ gamma,
                                 const float* __restrict__ beta,
                                 const _Float16* __restrict__ wf,
                                 float* __restrict__ out) {
  __shared__ __align__(32) char smem[65536];
  const int tid  = threadIdx.x;
  const int lane = tid & 31;
  const int wave = tid >> 5;
  const long rowBase = (long)blockIdx.x * TM;

  // gamma/beta for this lane's 8 columns (col j uses gamma[j-1]; col 0 unused)
  float gam[8], bet[8];
#pragma unroll
  for (int i = 0; i < 8; ++i) {
    int j = lane * 8 + i;
    gam[i] = (j == 0) ? 0.f : gamma[j - 1];
    bet[i] = (j == 0) ? 0.f : beta[j - 1];
  }

  // ---- Phase A: logmap0 + layernorm; (expmap0;logmap0) roundtrip == identity
  // after _proj, so x = [0, layernorm(...)]. Write f16 A-fragments to LDS.
  const int ks_a = lane >> 2;               // k-step of this lane's 8 cols
  const int hi_a = lane & 1;                // fragment lane-half select
  const int i0_a = ((lane >> 1) & 1) * 16;  // byte offset within 32B lane slot
  for (int it = 0; it < TM / 8; ++it) {
    int r8 = it * 8 + wave;
    const float* hp = H + (rowBase + r8) * DIM + lane * 8;
    float v[8];
#pragma unroll
    for (int i = 0; i < 8; ++i) v[i] = hp[i];
    float h0 = __shfl(v[0], 0, 32);
    float ss = 0.f;
#pragma unroll
    for (int i = 0; i < 8; ++i) ss += v[i] * v[i];
    if (lane == 0) ss -= v[0] * v[0];       // exclude component 0
    ss = wred(ss);
    float yn  = fmaxf(sqrtf(ss), 1e-15f);
    float th1 = acoshf(fmaxf(h0, 1.0f + 1e-7f));
    float sc  = th1 / yn;
    float t[8];
#pragma unroll
    for (int i = 0; i < 8; ++i) t[i] = v[i] * sc;
    if (lane == 0) t[0] = 0.f;
    float st = 0.f, st2 = 0.f;
#pragma unroll
    for (int i = 0; i < 8; ++i) { st += t[i]; st2 += t[i] * t[i]; }
    st = wred(st); st2 = wred(st2);
    float mu   = st * (1.0f / 255.0f);
    float var  = st2 * (1.0f / 255.0f) - mu * mu;
    float rstd = rsqrtf(var + 1e-5f);
    v8h pack;
#pragma unroll
    for (int i = 0; i < 8; ++i) {
      float x = (t[i] - mu) * rstd * gam[i] + bet[i];
      if (lane == 0 && i == 0) x = 0.f;
      pack[i] = (_Float16)x;
    }
    int mt = r8 >> 4;
    int lane_t = (r8 & 15) + (hi_a << 4);
    *(v8h*)(smem + (((mt * 8 + ks_a) * 32 + lane_t) * 32 + i0_a)) = pack;
  }
  __syncthreads();

  // ---- Phase B: GEMM via WMMA f16->f32. Each wave: 2 col-tiles x 4 row-tiles.
  v8f acc[8] = {};
  for (int jti = 0; jti < 2; ++jti) {
    int jt = wave * 2 + jti;
    for (int ks = 0; ks < 8; ++ks) {
      v16h bfrag = *(const v16h*)(wf + ((jt * 8 + ks) * 32 + lane) * 16);
#pragma unroll
      for (int mt = 0; mt < 4; ++mt) {
        v16h afrag = *(const v16h*)(smem + ((mt * 8 + ks) * 32 + lane) * 32);
        acc[jti * 4 + mt] = __builtin_amdgcn_wmma_f32_16x16x32_f16(
            false, afrag, false, bfrag, (short)0, acc[jti * 4 + mt],
            false, false);
      }
    }
  }
  __syncthreads();   // all A-fragment reads done; safe to reuse LDS as G

  float* G = (float*)smem;
#pragma unroll
  for (int t = 0; t < 8; ++t) {
    int jt  = wave * 2 + (t >> 2);
    int mt  = t & 3;
    int col = jt * 16 + (lane & 15);
    int rb  = mt * 16 + ((lane >> 4) << 3);
#pragma unroll
    for (int v = 0; v < 8; ++v) G[(rb + v) * DIM + col] = acc[t][v];
  }
  __syncthreads();

  // ---- Phase C: post-GEMM hyperbolic chain per row.
  float bv[8];
#pragma unroll
  for (int i = 0; i < 8; ++i)
    bv[i] = (lane == 0 && i == 0) ? 0.f : bias[lane * 8 + i];   // zero0(bias)

  for (int it = 0; it < TM / 8; ++it) {
    int r8 = it * 8 + wave;
    float g[8];
#pragma unroll
    for (int i = 0; i < 8; ++i) g[i] = G[r8 * DIM + lane * 8 + i];
    if (lane == 0) g[0] = 0.f;                                  // zero0(x)
    float ss = 0.f, sd = 0.f;
#pragma unroll
    for (int i = 0; i < 8; ++i) { ss += g[i] * g[i]; sd += g[i] * bv[i]; }
    ss = wred(ss); sd = wred(sd);
    // x2 = expmap0([0,g]):  x2_0 = cosh(n2), x2_j = sinh(n2)*g_j/n2
    float n2 = fmaxf(sqrtf(ss), 1e-15f);
    float inv_n2 = 1.f / n2;
    float sh2 = sinhf(n2), ch2 = coshf(n2);
    // transp0: yhat = g/n2; alpha = dot(yhat, bias)
    float alpha = sd * inv_n2;
    float cfac  = alpha * (1.0f - ch2) * inv_n2;
    float w[8];
#pragma unroll
    for (int i = 0; i < 8; ++i) w[i] = bv[i] - cfac * g[i];
    float s1 = 0.f, s2 = 0.f;
#pragma unroll
    for (int i = 0; i < 8; ++i) { s1 += g[i] * w[i]; s2 += w[i] * w[i]; }
    s1 = wred(s1); s2 = wred(s2);
    float ux = sh2 * inv_n2 * s1;                 // dot(x2[1:], w[1:])
    float b0 = ux / fmaxf(ch2, 1e-15f);           // proj_tan component 0
    // expmap(b, x2)
    float mdot  = s2 - b0 * b0;
    float normu = fminf(sqrtf(fmaxf(mdot, 0.f)), 1000.0f);
    float th    = fmaxf(normu, 1e-15f);
    float chT   = coshf(th);
    float shTt  = sinhf(th) / th;
    float x2s   = sh2 * inv_n2;
    float res[8]; float sr = 0.f;
#pragma unroll
    for (int i = 0; i < 8; ++i) {
      res[i] = chT * x2s * g[i] + shTt * w[i];
      sr += res[i] * res[i];
    }
    sr = wred(sr);
    float res0 = sqrtf(fmaxf(1.0f + sr, 0.004f)); // _proj component 0
    // relu(logmap0(res))
    float yn3 = fmaxf(sqrtf(sr), 1e-15f);
    float th3 = acoshf(fmaxf(res0, 1.0f + 1e-7f));
    float sc3 = th3 / yn3;
    float xt[8]; float sx = 0.f;
#pragma unroll
    for (int i = 0; i < 8; ++i) {
      xt[i] = fmaxf(res[i] * sc3, 0.f);
      sx += xt[i] * xt[i];
    }
    sx = wred(sx);
    // expmap0(zero0(xt))
    float n4  = fmaxf(sqrtf(sx), 1e-15f);
    float sc4 = sinhf(n4) / n4;
    float o0  = coshf(n4);
    float* op = out + (rowBase + r8) * DIM + lane * 8;
#pragma unroll
    for (int i = 0; i < 8; ++i) {
      float o = xt[i] * sc4;
      if (lane == 0 && i == 0) o = o0;
      op[i] = o;
    }
  }
}

extern "C" void kernel_launch(void* const* d_in, const int* in_sizes, int n_in,
                              void* d_out, int out_size, void* d_ws, size_t ws_size,
                              hipStream_t stream) {
  const float* H     = (const float*)d_in[0];
  const float* W     = (const float*)d_in[1];
  const float* bias  = (const float*)d_in[2];
  const float* gamma = (const float*)d_in[3];
  const float* beta  = (const float*)d_in[4];
  float* out = (float*)d_out;
  _Float16* wf = (_Float16*)d_ws;                 // 128 KB of workspace

  prep_w_kernel<<<256, 256, 0, stream>>>(W, wf);  // W -> f16 B-fragments
  int Nrows = in_sizes[0] / DIM;
  hnn_fused_kernel<<<Nrows / TM, 256, 0, stream>>>(H, bias, gamma, beta, wf, out);
}
